// InteractingAnimaleLayer_3925600109035
// MI455X (gfx1250) — compile-verified, matrix operand order checked by reading, not run on hardware
//
#include <hip/hip_runtime.h>

// Fused MHA for MI455X (gfx1250, wave32). B=1024, F=128, E=256, H=8, d=32.
// ~86 GFLOP vs ~257 MB min HBM traffic (~11us @ 23.3 TB/s) => near balance.
// 1 workgroup per batch element, 8 waves = 8 heads, all matmuls on
// v_wmma_f32_16x16x32_f16, X_b staged via Tensor Data Mover (double-buffered
// staging aliasing later-phase LDS), weights pre-converted once to a
// fragment-swizzled f16 workspace (L2-resident), non-temporal output stores.

typedef __attribute__((ext_vector_type(16))) _Float16 v16h;
typedef __attribute__((ext_vector_type(8)))  _Float16 v8h;
typedef __attribute__((ext_vector_type(4)))  _Float16 v4h;
typedef __attribute__((ext_vector_type(8)))  float    v8f;
typedef unsigned int u32x4 __attribute__((ext_vector_type(4)));
typedef int          i32x8 __attribute__((ext_vector_type(8)));
typedef int          i32x4 __attribute__((ext_vector_type(4)));

#define EDIM  256
#define FDIM  128
#define HEADS 8
#define DHEAD 32
#define WF_MAT 65536                    // halfs per swizzled matrix (8*16*32*16)
#define WF_BYTES (4u * WF_MAT * 2u)     // 512 KB workspace

#if defined(__HIP_DEVICE_COMPILE__) && __has_builtin(__builtin_amdgcn_tensor_load_to_lds)
#define USE_TDM 1
#else
#define USE_TDM 0
#endif

// A/B fragment (16x32 f16) from LDS with contiguous K: lane<16 -> row=lane,
// K {0..7,16..23}; lane>=16 -> row=lane-16, K {8..15,24..31}. 2x ds_load_b128.
__device__ __forceinline__ v16h frag_lds(const _Float16* base, int stride, int lane) {
  const int r    = lane & 15;
  const int koff = (lane < 16) ? 0 : 8;
  const _Float16* p = base + r * stride + koff;
  v8h lo = *(const v8h*)(p);
  v8h hi = *(const v8h*)(p + 16);
  v16h a;
#pragma unroll
  for (int e = 0; e < 8; ++e) { a[e] = lo[e]; a[8 + e] = hi[e]; }
  return a;
}

// B fragment (32x16) straight from global fp32 weight W[k][c] (fallback path).
__device__ __forceinline__ v16h frag_wgl(const float* __restrict__ W, int k0, int c0, int lane) {
  const int c  = c0 + (lane & 15);
  const int kb = (lane < 16) ? 0 : 8;
  v16h b;
#pragma unroll
  for (int e = 0; e < 16; ++e) {
    const int d = kb + (e < 8 ? e : e + 8);
    b[e] = (_Float16)W[(k0 + d) * EDIM + c];
  }
  return b;
}

// B fragment from the pre-swizzled f16 workspace: one contiguous 32B read.
template <bool PRE>
__device__ __forceinline__ v16h frag_b(const float* __restrict__ W,
                                       const _Float16* __restrict__ Wf,
                                       int kk, int c0, int lane) {
  if constexpr (PRE) {
    const _Float16* p = Wf + ((((kk << 4) + (c0 >> 4)) << 5) + lane) * 16;
    return *(const v16h*)p;          // two global_load_b128, no conversion
  } else {
    return frag_wgl(W, kk * 32, c0, lane);
  }
}

// C fragment (lane = N, vgpr v = row v / v+8) -> f16 row-major LDS tile.
__device__ __forceinline__ void store_c_rowmajor(_Float16* base, int stride, int lane, v8f c) {
  const int col = lane & 15;
  const int r0  = (lane < 16) ? 0 : 8;
#pragma unroll
  for (int v = 0; v < 8; ++v) base[(r0 + v) * stride + col] = (_Float16)c[v];
}

// C fragment -> transposed f16 LDS tile: one packed ds_store_b128 per lane.
__device__ __forceinline__ void store_c_transposed(_Float16* base, int stride, int lane, v8f c) {
  const int col = lane & 15;
  const int r0  = (lane < 16) ? 0 : 8;
  v8h h;
#pragma unroll
  for (int v = 0; v < 8; ++v) h[v] = (_Float16)c[v];
  *(v8h*)(base + col * stride + r0) = h;
}

__device__ __forceinline__ v8f wmma_f16(v16h a, v16h b, v8f c) {
  return __builtin_amdgcn_wmma_f32_16x16x32_f16(false, a, false, b, (short)0, c, false, false);
}

#if USE_TDM
__device__ __forceinline__ unsigned lds_addr_of(const void* p) {
  // Generic LDS pointers carry the aperture in [63:32]; [31:0] is the offset.
  return (unsigned)(unsigned long long)(size_t)p;
}

// Issue one TDM 2D tile load: rows x cols fp32, global -> LDS. D# per ISA ch.8.
// 6-arg builtin: (u32x4 g0, i32x8 g1, i32x4 g2, i32x4 g3, i32x8 pad, i32 cpol)
__device__ __forceinline__ void tdm_load_2d_f32(unsigned lds_off, const float* gsrc,
                                                unsigned rows, unsigned cols) {
  const unsigned long long ga = (unsigned long long)(size_t)gsrc;
  u32x4 g0;
  g0[0] = 1u;                                                 // count=1, user mode
  g0[1] = lds_off;                                            // lds_addr (bytes)
  g0[2] = (unsigned)ga;                                       // global_addr[31:0]
  g0[3] = (unsigned)((ga >> 32) & 0x01ffffffu) | (2u << 30);  // addr[56:32] | type=2
  i32x8 g1;
  g1[0] = 2 << 16;                              // data_size=4B, no multicast/pad
  g1[1] = (int)(cols << 16);                    // tensor_dim0[15:0]
  g1[2] = (int)((cols >> 16) | (rows << 16));   // tensor_dim0[31:16] | tensor_dim1[15:0]
  g1[3] = (int)((rows >> 16) | (cols << 16));   // tensor_dim1[31:16] | tile_dim0
  g1[4] = (int)rows;                            // tile_dim1 (tile_dim2 = 0)
  g1[5] = (int)cols;                            // tensor_dim0_stride[31:0]
  g1[6] = 0;
  g1[7] = 0;
  i32x4 z4 = {0, 0, 0, 0};                      // 2D: groups 2/3 unused
  i32x8 z8 = {0, 0, 0, 0, 0, 0, 0, 0};
  __builtin_amdgcn_tensor_load_to_lds(g0, g1, z4, z4, z8, 0);
}
#endif

// One-shot weight pre-pass: fp32 W[e][o] -> fragment-swizzled f16 workspace.
// Layout: Wf[m][kk][ctile][lane][e], so a B-frag is 32 contiguous bytes/lane.
__global__ __launch_bounds__(256)
void convert_weights_kernel(const float* __restrict__ Wq, const float* __restrict__ Wk,
                            const float* __restrict__ Wv, const float* __restrict__ Wr,
                            _Float16* __restrict__ Wf) {
  const int g    = blockIdx.x * 256 + threadIdx.x;   // 0 .. 4*65536-1
  const int m    = g >> 16;
  const int r    = g & 65535;
  const int kk   = r >> 13;
  const int ct   = (r >> 9) & 15;
  const int lane = (r >> 4) & 31;
  const int e    = r & 15;
  const float* W = (m == 0) ? Wq : (m == 1) ? Wk : (m == 2) ? Wv : Wr;
  const int c  = ct * 16 + (lane & 15);
  const int kb = (lane < 16) ? 0 : 8;
  const int d  = kb + (e < 8 ? e : e + 8);
  Wf[g] = (_Float16)W[(kk * 32 + d) * EDIM + c];
}

template <bool PRE>
__global__ __launch_bounds__(256, 1)
void mha_fused_kernel(const float* __restrict__ X,
                      const float* __restrict__ Wq,
                      const float* __restrict__ Wk,
                      const float* __restrict__ Wv,
                      const float* __restrict__ Wr,
                      const _Float16* __restrict__ Wf,
                      float* __restrict__ out) {
  // 288 KB LDS pool with phase-based aliasing (limit: 320 KB / workgroup).
  __shared__ __align__(16) char pool[294912];
  _Float16* Xs  = (_Float16*)(pool);            // [0,64K)    X_b f16 (phases 1-2b)
  _Float16* Qs  = (_Float16*)(pool + 65536);    // [64K,128K) Q row-major
  _Float16* Ks  = (_Float16*)(pool + 131072);   // [128K,192K) K row-major
  _Float16* Vt  = (_Float16*)(pool + 196608);   // [192K,256K) V transposed
  _Float16* Ps  = (_Float16*)(pool + 262144);   // [256K,288K) per-wave P scratch
  _Float16* Rs  = (_Float16*)(pool);            // aliases Xs after X is dead
  float*    Stg = (float*)(pool + 131072);      // 128KB fp32 staging, aliases Ks+Vt

  const int b    = blockIdx.x;
  const int tid  = threadIdx.x;
  const int lane = tid & 31;
  const int w    = tid >> 5;      // wave index == head index
  const int h32  = w * DHEAD;

  const _Float16* Wfq = Wf;
  const _Float16* Wfk = Wf + WF_MAT;
  const _Float16* Wfv = Wf + 2 * WF_MAT;
  const _Float16* Wfr = Wf + 3 * WF_MAT;

  // Hint the (L2-resident) weights into cache early.
  if constexpr (PRE) {
    __builtin_prefetch(Wf + tid * 1024, 0, 0);
  } else {
    __builtin_prefetch(Wq + tid * 256, 0, 0);
    __builtin_prefetch(Wk + tid * 256, 0, 0);
    __builtin_prefetch(Wv + tid * 256, 0, 0);
    __builtin_prefetch(Wr + tid * 256, 0, 0);
  }

  const float* Xg = X + (size_t)b * FDIM * EDIM;

  // ---- Phase 1: stage X_b. TDM double-buffer: DMA chunk1 overlaps convert0 ----
#if USE_TDM
  if (w == 0) {
    tdm_load_2d_f32(lds_addr_of(Stg),             Xg,             64u, (unsigned)EDIM);
    tdm_load_2d_f32(lds_addr_of(Stg + 64 * EDIM), Xg + 64 * EDIM, 64u, (unsigned)EDIM);
    __builtin_amdgcn_s_wait_tensorcnt(1);   // chunk0 landed
  }
  __syncthreads();
#pragma unroll 4
  for (int it = 0; it < 16; ++it) {         // convert rows 0..63 -> f16
    const int idx = it * 256 + tid;
    float4 x = ((const float4*)Stg)[idx];
    v4h hx;
    hx[0] = (_Float16)x.x; hx[1] = (_Float16)x.y;
    hx[2] = (_Float16)x.z; hx[3] = (_Float16)x.w;
    *(v4h*)(Xs + idx * 4) = hx;
  }
  if (w == 0) __builtin_amdgcn_s_wait_tensorcnt(0);
  __syncthreads();
#pragma unroll 4
  for (int it = 0; it < 16; ++it) {         // convert rows 64..127 -> f16
    const int idx = it * 256 + tid;
    float4 x = ((const float4*)(Stg + 64 * EDIM))[idx];
    v4h hx;
    hx[0] = (_Float16)x.x; hx[1] = (_Float16)x.y;
    hx[2] = (_Float16)x.z; hx[3] = (_Float16)x.w;
    *(v4h*)(Xs + 64 * EDIM + idx * 4) = hx;
  }
#else
#pragma unroll 4
  for (int it = 0; it < (FDIM * EDIM / 4) / 256; ++it) {
    const int idx = it * 256 + tid;
    float4 x = ((const float4*)Xg)[idx];
    v4h hx;
    hx[0] = (_Float16)x.x; hx[1] = (_Float16)x.y;
    hx[2] = (_Float16)x.z; hx[3] = (_Float16)x.w;
    *(v4h*)(Xs + idx * 4) = hx;
  }
#endif
  __syncthreads();   // staging dead; Ks/Vt/Qs regions may now be written

  // ---- Phase 2: K, V, Q projections. B-frags hoisted over the 8 row-tiles ----
#pragma unroll
  for (int jc = 0; jc < 2; ++jc) {
    const int c0 = h32 + jc * 16;
    v8f kacc[8] = {}; v8f vacc[8] = {}; v8f qacc[8] = {};
#pragma unroll 1
    for (int kk = 0; kk < 8; ++kk) {
      v16h bk = frag_b<PRE>(Wk, Wfk, kk, c0, lane);
      v16h bv = frag_b<PRE>(Wv, Wfv, kk, c0, lane);
      v16h bq = frag_b<PRE>(Wq, Wfq, kk, c0, lane);
#pragma unroll
      for (int i = 0; i < 8; ++i) {
        v16h a = frag_lds(Xs + (i * 16) * EDIM + kk * 32, EDIM, lane);
        kacc[i] = wmma_f16(a, bk, kacc[i]);
        vacc[i] = wmma_f16(a, bv, vacc[i]);
        qacc[i] = wmma_f16(a, bq, qacc[i]);
      }
    }
#pragma unroll
    for (int i = 0; i < 8; ++i) {
      store_c_rowmajor (Ks + (i * 16) * EDIM + c0, EDIM, lane, kacc[i]);
      store_c_transposed(Vt + c0 * FDIM + i * 16,  FDIM, lane, vacc[i]);
      store_c_rowmajor (Qs + (i * 16) * EDIM + c0, EDIM, lane, qacc[i]);
    }
  }

  // ---- Phase 2b: residual X@Wr into registers (X still live) ----
  v8f rs0[8] = {}; v8f rs1[8] = {};
#pragma unroll 1
  for (int kk = 0; kk < 8; ++kk) {
    v16h b0 = frag_b<PRE>(Wr, Wfr, kk, h32,      lane);
    v16h b1 = frag_b<PRE>(Wr, Wfr, kk, h32 + 16, lane);
#pragma unroll
    for (int i = 0; i < 8; ++i) {
      v16h a = frag_lds(Xs + (i * 16) * EDIM + kk * 32, EDIM, lane);
      rs0[i] = wmma_f16(a, b0, rs0[i]);
      rs1[i] = wmma_f16(a, b1, rs1[i]);
    }
  }
  __syncthreads();   // all waves done with Xs; its region becomes Rs
  _Float16* Rw = Rs + w * (FDIM * DHEAD);
#pragma unroll
  for (int i = 0; i < 8; ++i) {
    store_c_rowmajor(Rw + (i * 16) * DHEAD,      DHEAD, lane, rs0[i]);
    store_c_rowmajor(Rw + (i * 16) * DHEAD + 16, DHEAD, lane, rs1[i]);
  }

  // ---- Phase 3: per 16-row block: S = Q K^T, softmax, O = P V, epilogue ----
  _Float16* Pw = Ps + w * (16 * FDIM);
#pragma unroll 1
  for (int i = 0; i < 8; ++i) {
    v16h aq = frag_lds(Qs + (i * 16) * EDIM + h32, EDIM, lane);   // d = 32 = one K-step

    v8f S[8];
#pragma unroll
    for (int j = 0; j < 8; ++j) {
      v8f z = {};
      S[j] = wmma_f16(aq, frag_lds(Ks + (j * 16) * EDIM + h32, EDIM, lane), z);
    }

    // Row softmax; C-frag layout: vgpr v = row (v|v+8), lane%16 = column.
    float m[8];
#pragma unroll
    for (int v = 0; v < 8; ++v) {
      float mv = S[0][v];
#pragma unroll
      for (int j = 1; j < 8; ++j) mv = fmaxf(mv, S[j][v]);
#pragma unroll
      for (int off = 8; off >= 1; off >>= 1) mv = fmaxf(mv, __shfl_xor(mv, off, 32));
      m[v] = mv;
    }
    float sum[8];
#pragma unroll
    for (int v = 0; v < 8; ++v) sum[v] = 0.0f;
#pragma unroll
    for (int j = 0; j < 8; ++j) {
      v8f p;
#pragma unroll
      for (int v = 0; v < 8; ++v) {
        const float e = __expf(S[j][v] - m[v]);
        p[v] = e;
        sum[v] += e;
      }
      store_c_rowmajor(Pw + j * 16, FDIM, lane, p);   // unnormalized P (f16)
    }
#pragma unroll
    for (int v = 0; v < 8; ++v) {
#pragma unroll
      for (int off = 8; off >= 1; off >>= 1) sum[v] += __shfl_xor(sum[v], off, 32);
    }

    // O = P V (A from P scratch, B from transposed V; both contiguous-K).
    v8f o0 = {}, o1 = {};
#pragma unroll 1
    for (int kk = 0; kk < 4; ++kk) {
      v16h ap = frag_lds(Pw + kk * 32, FDIM, lane);
      o0 = wmma_f16(ap, frag_lds(Vt + h32 * FDIM + kk * 32,        FDIM, lane), o0);
      o1 = wmma_f16(ap, frag_lds(Vt + (h32 + 16) * FDIM + kk * 32, FDIM, lane), o1);
    }

    // Epilogue: relu(O/rowsum + Res) -> fp32 global. Non-temporal: write-once.
    const int col   = lane & 15;
    const int rbase = i * 16 + ((lane < 16) ? 0 : 8);
    float* og = out + ((size_t)b * FDIM + rbase) * EDIM + h32 + col;
#pragma unroll
    for (int v = 0; v < 8; ++v) {
      const float inv = 1.0f / sum[v];
      const float r0v = (float)Rw[(rbase + v) * DHEAD + col];
      const float r1v = (float)Rw[(rbase + v) * DHEAD + 16 + col];
      __builtin_nontemporal_store(fmaxf(o0[v] * inv + r0v, 0.0f), og + v * EDIM);
      __builtin_nontemporal_store(fmaxf(o1[v] * inv + r1v, 0.0f), og + v * EDIM + 16);
    }
  }
}

extern "C" void kernel_launch(void* const* d_in, const int* in_sizes, int n_in,
                              void* d_out, int out_size, void* d_ws, size_t ws_size,
                              hipStream_t stream) {
  (void)in_sizes; (void)n_in; (void)out_size;
  const float* X  = (const float*)d_in[0];
  const float* Wq = (const float*)d_in[1];
  const float* Wk = (const float*)d_in[2];
  const float* Wv = (const float*)d_in[3];
  const float* Wr = (const float*)d_in[4];
  float* out = (float*)d_out;
  dim3 grid(1024, 1, 1), block(256, 1, 1);
  if (ws_size >= (size_t)WF_BYTES) {
    _Float16* Wf = (_Float16*)d_ws;
    hipLaunchKernelGGL(convert_weights_kernel, dim3(1024, 1, 1), block, 0, stream,
                       Wq, Wk, Wv, Wr, Wf);
    hipLaunchKernelGGL(mha_fused_kernel<true>, grid, block, 0, stream,
                       X, Wq, Wk, Wv, Wr, (const _Float16*)Wf, out);
  } else {
    hipLaunchKernelGGL(mha_fused_kernel<false>, grid, block, 0, stream,
                       X, Wq, Wk, Wv, Wr, (const _Float16*)nullptr, out);
  }
}
